// LSTM_gat_91036126806231
// MI455X (gfx1250) — matrix operand order repaired and checked by reading, not compiled, not run on hardware
//
#include <hip/hip_runtime.h>
#include <hip/hip_bf16.h>

#define B_   16
#define T_   256
#define M_   10
#define K_   7
#define KR_  8
#define N_   80
#define BN_  1280
#define EMB  64
#define HID  128
#define G4   512
#define NEG_SLOPE 0.2f
#define EMAX 512

typedef __attribute__((ext_vector_type(16))) __bf16 bf16x16;
typedef __attribute__((ext_vector_type(8)))  __bf16 bf16x8;
typedef __attribute__((ext_vector_type(8)))  float  f32x8;

// ---------------------------------------------------------------- helpers
static __device__ inline bf16x16 ld16(const __bf16* p0, const __bf16* p1) {
    bf16x8 lo = *(const bf16x8*)p0;
    bf16x8 hi = *(const bf16x8*)p1;
    bf16x16 r;
#pragma unroll
    for (int i = 0; i < 8; ++i) { r[i] = lo[i]; r[i + 8] = hi[i]; }
    return r;
}

static __device__ inline f32x8 wmma_bf16(bf16x16 a, bf16x16 b, f32x8 c) {
    // D = A(16x32 bf16) * B(32x16 bf16) + C(16x16 f32)
    return __builtin_amdgcn_wmma_f32_16x16x32_bf16(
        false, a, false, b, (short)0, c, false, false);
}

static __device__ inline float sigmoidf_(float x) {
    return 1.0f / (1.0f + __expf(-x));
}
static __device__ inline float tanhf_(float x) {
    return 2.0f / (1.0f + __expf(-2.0f * x)) - 1.0f;
}

// ---------------------------------------------------------------- kernel 0:
// convert weights to bf16 (w_ih, w_hh, gat_w transposed), fuse biases
__global__ __launch_bounds__(256) void prep_kernel(
    const float* __restrict__ wih, const float* __restrict__ whh,
    const float* __restrict__ gw,  const float* __restrict__ bih,
    const float* __restrict__ bhh,
    unsigned short* __restrict__ wih_b, unsigned short* __restrict__ whh_b,
    unsigned short* __restrict__ gwt_b, float* __restrict__ bias)
{
    int id = blockIdx.x * 256 + threadIdx.x;
    if (id < G4 * EMB) ((__bf16*)wih_b)[id] = (__bf16)wih[id];
    if (id < G4 * HID) ((__bf16*)whh_b)[id] = (__bf16)whh[id];
    if (id < HID * HID) {
        int j = id / HID, k = id % HID;                 // gwt[j][k] = gw[k][j]
        ((__bf16*)gwt_b)[id] = (__bf16)gw[k * HID + j];
    }
    if (id < G4) bias[id] = bih[id] + bhh[id];
}

// ---------------------------------------------------------------- kernel 1:
// virtual root + temporal delta + phi MLP -> v (bf16, layout [t][b*N+node][64])
static __device__ inline void root_at(const float* __restrict__ pos,
                                      int b, int t, int m,
                                      float& r0, float& r1)
{
    float s0 = 0.f, s1 = 0.f, ws = 0.f;
#pragma unroll
    for (int k = 0; k < K_; ++k) {
        size_t i = ((((size_t)b * T_ + t) * M_ + m) * K_ + k) * 2;
        float a0 = pos[i], a1 = pos[i + 1];
        int f0 = isfinite(a0), f1 = isfinite(a1);
        float c0 = f0 ? a0 : 0.f, c1 = f1 ? a1 : 0.f;
        float w = (f0 && f1) ? 1.f : 0.f;
        s0 += c0 * w; s1 += c1 * w; ws += w;
    }
    float den = ws > 1.f ? ws : 1.f;
    r0 = s0 / den; r1 = s1 / den;
}

__global__ __launch_bounds__(256) void phi_kernel(
    const float* __restrict__ pos,
    const float* __restrict__ w1, const float* __restrict__ b1,
    const float* __restrict__ w2, const float* __restrict__ b2,
    unsigned short* __restrict__ vout_u)
{
    __bf16* vout = (__bf16*)vout_u;
    int id = blockIdx.x * 256 + threadIdx.x;
    if (id >= B_ * T_ * N_) return;
    int node = id % N_;
    int bt   = id / N_;
    int t    = bt % T_;
    int b    = bt / T_;
    int m    = node / KR_;
    int kp   = node % KR_;

    float d0 = 0.f, d1 = 0.f;
    if (t > 0) {
        if (kp < K_) {
            size_t i1 = ((((size_t)b * T_ + t)     * M_ + m) * K_ + kp) * 2;
            size_t i0 = ((((size_t)b * T_ + (t-1)) * M_ + m) * K_ + kp) * 2;
            float a0 = pos[i1],     a1 = pos[i1 + 1];
            float p0 = pos[i0],     p1 = pos[i0 + 1];
            a0 = isfinite(a0) ? a0 : 0.f;  a1 = isfinite(a1) ? a1 : 0.f;
            p0 = isfinite(p0) ? p0 : 0.f;  p1 = isfinite(p1) ? p1 : 0.f;
            d0 = a0 - p0;  d1 = a1 - p1;
        } else {
            float r0t, r1t, r0p, r1p;
            root_at(pos, b, t,     m, r0t, r1t);
            root_at(pos, b, t - 1, m, r0p, r1p);
            d0 = r0t - r0p;  d1 = r1t - r1p;
        }
    }

    float h1[EMB];
#pragma unroll
    for (int j = 0; j < EMB; ++j) {
        float x = d0 * w1[j] + d1 * w1[EMB + j] + b1[j];
        h1[j] = x > 0.f ? x : 0.f;
    }
    float o[EMB];
#pragma unroll
    for (int j = 0; j < EMB; ++j) o[j] = b2[j];
    for (int i = 0; i < EMB; ++i) {
        float hv = h1[i];
#pragma unroll
        for (int j = 0; j < EMB; ++j) o[j] += hv * w2[i * EMB + j];
    }

    __bf16* dst = vout + ((size_t)t * BN_ + (size_t)b * N_ + node) * EMB;
#pragma unroll
    for (int j0 = 0; j0 < EMB; j0 += 8) {
        bf16x8 pk;
#pragma unroll
        for (int j = 0; j < 8; ++j) pk[j] = (__bf16)o[j0 + j];
        *(bf16x8*)(dst + j0) = pk;
    }
}

// ---------------------------------------------------------------- kernel 2:
// LSTM over T. One WG per 16-row tile of BN; 8 waves, wave w owns hidden
// cols [16w,16w+16) and its i/f/g/o gate tiles. Weights register-resident.
__global__ __launch_bounds__(256, 1) void lstm_kernel(
    const unsigned short* __restrict__ vbf_u,
    const unsigned short* __restrict__ wih_u,
    const unsigned short* __restrict__ whh_u,
    const float* __restrict__ bias,
    float* __restrict__ out_r)
{
    const __bf16* vbf = (const __bf16*)vbf_u;
    const __bf16* wih = (const __bf16*)wih_u;
    const __bf16* whh = (const __bf16*)whh_u;

    __shared__ __bf16 sh[16 * HID];     // current h tile (bf16)

    const int tid  = threadIdx.x;
    const int wave = tid >> 5;
    const int lane = tid & 31;
    const int n16  = lane & 15;
    const int half = lane >> 4;

    const int row0  = blockIdx.x * 16;
    const int b     = row0 / N_;        // 80 % 16 == 0 -> no b straddle
    const int node0 = row0 % N_;
    const int col   = 16 * wave + n16;

    // B fragments: Wih^T / Whh^T columns for this wave's 4 gate tiles
    bf16x16 Bx[4][2];
    bf16x16 Bh[4][4];
#pragma unroll
    for (int g = 0; g < 4; ++g) {
        const int wrow = g * HID + col;
#pragma unroll
        for (int ks = 0; ks < 2; ++ks) {
            const __bf16* p = wih + (size_t)wrow * EMB + 32 * ks + 8 * half;
            Bx[g][ks] = ld16(p, p + 16);
        }
#pragma unroll
        for (int ks = 0; ks < 4; ++ks) {
            const __bf16* p = whh + (size_t)wrow * HID + 32 * ks + 8 * half;
            Bh[g][ks] = ld16(p, p + 16);
        }
    }
    float bg[4];
#pragma unroll
    for (int g = 0; g < 4; ++g) bg[g] = bias[g * HID + col];

    f32x8 cst = {0.f, 0.f, 0.f, 0.f, 0.f, 0.f, 0.f, 0.f};

    for (int i = tid; i < 16 * HID; i += 256) sh[i] = (__bf16)0.f;
    __syncthreads();

    for (int t = 0; t < T_; ++t) {
        // A fragments: x from global, h from LDS (row = lane%16)
        const __bf16* xb = vbf + ((size_t)t * BN_ + row0 + n16) * EMB;
        bf16x16 Ax[2];
#pragma unroll
        for (int ks = 0; ks < 2; ++ks) {
            const __bf16* p = xb + 32 * ks + 8 * half;
            Ax[ks] = ld16(p, p + 16);
        }
        bf16x16 Ah[4];
#pragma unroll
        for (int ks = 0; ks < 4; ++ks) {
            const __bf16* p = sh + n16 * HID + 32 * ks + 8 * half;
            Ah[ks] = ld16(p, p + 16);
        }

        f32x8 acc[4];
#pragma unroll
        for (int g = 0; g < 4; ++g)
#pragma unroll
            for (int r = 0; r < 8; ++r) acc[g][r] = bg[g];

#pragma unroll
        for (int g = 0; g < 4; ++g) {
#pragma unroll
            for (int ks = 0; ks < 2; ++ks)
                acc[g] = wmma_bf16(Ax[ks], Bx[g][ks], acc[g]);
#pragma unroll
            for (int ks = 0; ks < 4; ++ks)
                acc[g] = wmma_bf16(Ah[ks], Bh[g][ks], acc[g]);
        }

        __syncthreads();   // everyone done reading sh

#pragma unroll
        for (int r = 0; r < 8; ++r) {
            float iv = sigmoidf_(acc[0][r]);
            float fv = sigmoidf_(acc[1][r]);
            float gv = tanhf_(acc[2][r]);
            float ov = sigmoidf_(acc[3][r]);
            float cn = fv * cst[r] + iv * gv;
            cst[r] = cn;
            float hv = ov * tanhf_(cn);
            int m = r + 8 * half;
            out_r[(((size_t)b * T_ + t) * N_ + node0 + m) * HID + col] = hv;
            sh[m * HID + col] = (__bf16)hv;
        }
        __syncthreads();
    }
}

// ---------------------------------------------------------------- kernel 3:
// GAT: one WG per (b,t). h = r*W via WMMA into LDS, then CSR softmax/gather.
__global__ __launch_bounds__(256, 1) void gat_kernel(
    const float* __restrict__ r_in,
    const unsigned short* __restrict__ gwt_u,
    const float* __restrict__ a_src,
    const float* __restrict__ a_dst,
    const int* __restrict__ eidx, int E,
    float* __restrict__ out_rhat)
{
    const __bf16* gwt = (const __bf16*)gwt_u;
    __shared__ float hgat[N_ * HID];        // 40KB
    __shared__ float ssrc[N_], sdst[N_], ssum[N_];
    __shared__ float alpha[EMAX];
    __shared__ int   esrc[EMAX], edst[EMAX];
    __shared__ int   cnt[N_], offs[N_ + 1], cur[N_], elist[EMAX];

    const int tid  = threadIdx.x;
    const int wave = tid >> 5;
    const int lane = tid & 31;
    const int n16  = lane & 15;
    const int half = lane >> 4;
    const int bt   = blockIdx.x;
    const float* rb = r_in + (size_t)bt * N_ * HID;

    // B fragments of gat_w (via pre-transposed bf16 copy)
    bf16x16 Bw[4];
#pragma unroll
    for (int ks = 0; ks < 4; ++ks) {
        const __bf16* p = gwt + (size_t)(16 * wave + n16) * HID
                              + 32 * ks + 8 * half;
        Bw[ks] = ld16(p, p + 16);
    }

#pragma unroll
    for (int rt = 0; rt < 5; ++rt) {
        f32x8 acc = {0.f, 0.f, 0.f, 0.f, 0.f, 0.f, 0.f, 0.f};
#pragma unroll
        for (int ks = 0; ks < 4; ++ks) {
            const float* ap = rb + (size_t)(16 * rt + n16) * HID
                                 + 32 * ks + 8 * half;
            bf16x16 a;
#pragma unroll
            for (int i = 0; i < 8; ++i) {
                a[i]     = (__bf16)ap[i];
                a[i + 8] = (__bf16)ap[16 + i];
            }
            acc = wmma_bf16(a, Bw[ks], acc);
        }
#pragma unroll
        for (int r = 0; r < 8; ++r)
            hgat[(16 * rt + r + 8 * half) * HID + 16 * wave + n16] = acc[r];
    }
    __syncthreads();

    // attention scores + edge load + CSR count init
    if (tid < 2 * N_) {
        int n = tid % N_;
        const float* av = (tid < N_) ? a_src : a_dst;
        float s = 0.f;
        for (int k = 0; k < HID; ++k) s += hgat[n * HID + k] * av[k];
        if (tid < N_) ssrc[n] = s; else sdst[n] = s;
    }
    if (tid < E) { esrc[tid] = eidx[tid]; edst[tid] = eidx[E + tid]; }
    if (tid < N_) cnt[tid] = 0;
    __syncthreads();
    if (tid < E) atomicAdd(&cnt[edst[tid]], 1);
    __syncthreads();
    if (tid == 0) {
        int o = 0;
        for (int d = 0; d < N_; ++d) { offs[d] = o; o += cnt[d]; }
        offs[N_] = o;
    }
    __syncthreads();
    if (tid < N_) cur[tid] = offs[tid];
    __syncthreads();
    if (tid < E) { int p = atomicAdd(&cur[edst[tid]], 1); elist[p] = tid; }
    __syncthreads();

    // per-dst stable softmax (each dst owned by one thread)
    if (tid < N_) {
        int d = tid;
        float mx = -1e30f;
        for (int j = offs[d]; j < offs[d + 1]; ++j) {
            int e = elist[j];
            float s = ssrc[esrc[e]] + sdst[d];
            s = s > 0.f ? s : NEG_SLOPE * s;
            mx = fmaxf(mx, s);
        }
        float sum = 0.f;
        for (int j = offs[d]; j < offs[d + 1]; ++j) {
            int e = elist[j];
            float s = ssrc[esrc[e]] + sdst[d];
            s = s > 0.f ? s : NEG_SLOPE * s;
            float w = __expf(s - mx);
            alpha[e] = w;       // each edge has a unique dst -> no race
            sum += w;
        }
        ssum[d] = sum;
    }
    __syncthreads();
    if (tid < E) alpha[tid] = alpha[tid] / ssum[edst[tid]];
    __syncthreads();

    // gather + ELU + store
    for (int idx = tid; idx < N_ * HID; idx += 256) {
        int d = idx >> 7, c = idx & (HID - 1);
        float acc = 0.f;
        for (int j = offs[d]; j < offs[d + 1]; ++j) {
            int e = elist[j];
            acc += alpha[e] * hgat[esrc[e] * HID + c];
        }
        float y = acc > 0.f ? acc : (__expf(acc) - 1.f);
        out_rhat[(size_t)bt * (N_ * HID) + idx] = y;
    }
}

// ---------------------------------------------------------------- launch
extern "C" void kernel_launch(void* const* d_in, const int* in_sizes, int n_in,
                              void* d_out, int out_size, void* d_ws, size_t ws_size,
                              hipStream_t stream) {
    const float* pos    = (const float*)d_in[0];
    const float* phi_w1 = (const float*)d_in[1];
    const float* phi_b1 = (const float*)d_in[2];
    const float* phi_w2 = (const float*)d_in[3];
    const float* phi_b2 = (const float*)d_in[4];
    const float* w_ih   = (const float*)d_in[5];
    const float* w_hh   = (const float*)d_in[6];
    const float* b_ih   = (const float*)d_in[7];
    const float* b_hh   = (const float*)d_in[8];
    const float* gat_w  = (const float*)d_in[9];
    const float* a_src  = (const float*)d_in[10];
    const float* a_dst  = (const float*)d_in[11];
    const int*   edges  = (const int*)d_in[12];
    int E = in_sizes[12] / 2;

    char* ws = (char*)d_ws;
    size_t off = 0;
    unsigned short* vbf    = (unsigned short*)(ws + off); off += (size_t)T_ * BN_ * EMB * 2;
    unsigned short* wih_b  = (unsigned short*)(ws + off); off += (size_t)G4 * EMB * 2;
    unsigned short* whh_b  = (unsigned short*)(ws + off); off += (size_t)G4 * HID * 2;
    unsigned short* gwt_b  = (unsigned short*)(ws + off); off += (size_t)HID * HID * 2;
    float*          bias   = (float*)(ws + off);          off += (size_t)G4 * 4;

    float* out_r    = (float*)d_out;
    float* out_rhat = out_r + (size_t)B_ * T_ * N_ * HID;

    prep_kernel<<<(G4 * HID + 255) / 256, 256, 0, stream>>>(
        w_ih, w_hh, gat_w, b_ih, b_hh, wih_b, whh_b, gwt_b, bias);

    phi_kernel<<<(B_ * T_ * N_ + 255) / 256, 256, 0, stream>>>(
        pos, phi_w1, phi_b1, phi_w2, phi_b2, vbf);

    lstm_kernel<<<BN_ / 16, 256, 0, stream>>>(
        vbf, wih_b, whh_b, bias, out_r);

    gat_kernel<<<B_ * T_, 256, 0, stream>>>(
        out_r, gwt_b, a_src, a_dst, edges, E, out_rhat);
}
